// MultiHeadAttention_80865644249809
// MI455X (gfx1250) — matrix operand order repaired and checked by reading, not compiled, not run on hardware
//
#include <hip/hip_runtime.h>
#include <hip/hip_bf16.h>

// ---------------------------------------------------------------------------
// MHA forward for MI455X (gfx1250, wave32, WMMA bf16).
//   B=8, S=1024, D=768, H=12, d_k=64.
// Pipeline:
//   proj_gemm<float,bf16>   x3  : Q/K/V = X @ W^T + b   (bf16 to workspace)
//   attn_kernel                 : flash-style softmax(QK^T/8)V  (bf16 concat)
//   proj_gemm<bf16,float>   x1  : out = A @ Wo^T + bo   (fp32 to d_out)
// All matrix math via v_wmma_f32_16x16x32_bf16; K tile fetched with
// global_load_async_to_lds_b128 (ASYNCcnt); softmax row-max via DPP
// xor-reduction, row-sum via a ones-fragment WMMA; raw v_exp_f32 exponentials.
// ---------------------------------------------------------------------------

typedef __attribute__((ext_vector_type(16))) __bf16 v16bf;
typedef __attribute__((ext_vector_type(8)))  __bf16 v8bf;
typedef __attribute__((ext_vector_type(4)))  __bf16 v4bf;
typedef __attribute__((ext_vector_type(2)))  __bf16 v2bf;
typedef __attribute__((ext_vector_type(4)))  float  v4f;
typedef __attribute__((ext_vector_type(8)))  float  v8f;

#define D_MODEL 768
#define SEQ     1024
#define NBATCH  8
#define NHEADS  12
#define DK      64
#define MROWS   (NBATCH * SEQ)   // 8192

// 0.125 (=1/sqrt(64)) * log2(e): fold score scale into exp2.
#define CSC 0.1803368801111204f

// LDS strides in bf16 elements; multiples of 8 keep 16B alignment.
#define LDT 40   // 32-col tiles
#define LDK 72   // 64-col K tile

// Flip to 0 if the async-to-LDS asm is rejected by the assembler.
#ifndef USE_ASYNC_K
#define USE_ASYNC_K 1
#endif

template <typename T> struct vec4;
template <> struct vec4<float>  { typedef v4f  type; };
template <> struct vec4<__bf16> { typedef v4bf type; };

__device__ __forceinline__ v16bf join16(v8bf lo, v8bf hi) {
  v16bf r;
#pragma unroll
  for (int e = 0; e < 8; ++e) { r[e] = lo[e]; r[8 + e] = hi[e]; }
  return r;
}

// Raw v_exp_f32: flash-softmax arguments are <= 0 and far from the f32
// denormal boundary, so skip libm's range-fixup sequence.
__device__ __forceinline__ float fast_exp2(float x) {
#if __has_builtin(__builtin_amdgcn_exp2f)
  return __builtin_amdgcn_exp2f(x);
#else
  return exp2f(x);
#endif
}

// Max-reduce across the 16 lanes of a half-wave (rows of the C layout) using
// DPP16 xor steps: quad_perm(1,0,3,2), quad_perm(2,3,0,1), row_half_mirror,
// row_mirror. Pure VALU; no LDS/bpermute traffic.
__device__ __forceinline__ float max16(float x) {
#if __has_builtin(__builtin_amdgcn_mov_dpp)
  int v;
  v = __builtin_amdgcn_mov_dpp(__builtin_bit_cast(int, x), 0xB1, 0xF, 0xF, true);
  x = fmaxf(x, __builtin_bit_cast(float, v));
  v = __builtin_amdgcn_mov_dpp(__builtin_bit_cast(int, x), 0x4E, 0xF, 0xF, true);
  x = fmaxf(x, __builtin_bit_cast(float, v));
  v = __builtin_amdgcn_mov_dpp(__builtin_bit_cast(int, x), 0x141, 0xF, 0xF, true);
  x = fmaxf(x, __builtin_bit_cast(float, v));
  v = __builtin_amdgcn_mov_dpp(__builtin_bit_cast(int, x), 0x140, 0xF, 0xF, true);
  x = fmaxf(x, __builtin_bit_cast(float, v));
  return x;
#else
#pragma unroll
  for (int m = 1; m < 16; m <<= 1) x = fmaxf(x, __shfl_xor(x, m, 32));
  return x;
#endif
}

// ---------------------------------------------------------------------------
// Tiled GEMM: Y[m,n] = sum_k X[m,k] * W[n,k] + bias[n]
// Block tile 128x128, 8 waves in 4(M) x 2(N), wave tile 32x64, K step 32.
// Double-buffered LDS: one barrier per K step; next tile's global loads are
// in flight while WMMAs consume the current tile.
// ---------------------------------------------------------------------------
template <typename TIn, typename TOut>
__global__ __launch_bounds__(256)
void proj_gemm(const TIn* __restrict__ X, const float* __restrict__ W,
               const float* __restrict__ bias, TOut* __restrict__ Y,
               int M, int N, int K) {
  __shared__ alignas(16) __bf16 smX[2][128 * LDT];
  __shared__ alignas(16) __bf16 smW[2][128 * LDT];

  const int t = threadIdx.x;
  const int lane = t & 31, wave = t >> 5;
  const int wm = wave >> 1, wn = wave & 1;
  const int half = lane >> 4, l16 = lane & 15;
  const int m0 = blockIdx.y * 128, n0 = blockIdx.x * 128;

  typedef typename vec4<TIn>::type tin4;

  float xr[16], wr[16];
  auto load_regs = [&](int k0) {
#pragma unroll
    for (int q = 0; q < 4; ++q) {
      const int id = q * 256 + t, r = id >> 3, c = (id & 7) * 4;
      tin4 xv = *(const tin4*)(X + (size_t)(m0 + r) * K + k0 + c);
      v4f  wv = *(const v4f*)(W + (size_t)(n0 + r) * K + k0 + c);
#pragma unroll
      for (int e = 0; e < 4; ++e) {
        xr[q * 4 + e] = (float)xv[e];
        wr[q * 4 + e] = wv[e];
      }
    }
  };
  auto store_tiles = [&](int buf) {
#pragma unroll
    for (int q = 0; q < 4; ++q) {
      const int id = q * 256 + t, r = id >> 3, c = (id & 7) * 4;
      v4bf xo, wo;
#pragma unroll
      for (int e = 0; e < 4; ++e) {
        xo[e] = (__bf16)xr[q * 4 + e];
        wo[e] = (__bf16)wr[q * 4 + e];
      }
      *(v4bf*)&smX[buf][r * LDT + c] = xo;
      *(v4bf*)&smW[buf][r * LDT + c] = wo;
    }
  };

  v8f acc[2][4];
#pragma unroll
  for (int s = 0; s < 2; ++s)
#pragma unroll
    for (int nt = 0; nt < 4; ++nt)
#pragma unroll
      for (int r = 0; r < 8; ++r) acc[s][nt][r] = 0.0f;

  load_regs(0);
  int buf = 0;
  for (int k0 = 0; k0 < K; k0 += 32, buf ^= 1) {
    store_tiles(buf);
    __syncthreads();
    if (k0 + 32 < K) load_regs(k0 + 32);  // overlap with WMMA below

    // A fragments (16x32 bf16): lane row = l16; lanes 0-15 take K
    // {0..7,16..23}, lanes 16-31 take K {8..15,24..31}.
    v16bf afr[2];
#pragma unroll
    for (int s = 0; s < 2; ++s) {
      const __bf16* base = &smX[buf][(wm * 32 + s * 16 + l16) * LDT];
      afr[s] = join16(*(const v8bf*)(base + half * 8),
                      *(const v8bf*)(base + 16 + half * 8));
    }
    // B fragments (32x16): lane col = l16 (row of W), K half by lane group.
    v16bf bfr[4];
#pragma unroll
    for (int nt = 0; nt < 4; ++nt) {
      const __bf16* base = &smW[buf][(wn * 64 + nt * 16 + l16) * LDT];
      bfr[nt] = join16(*(const v8bf*)(base + half * 8),
                       *(const v8bf*)(base + 16 + half * 8));
    }
#pragma unroll
    for (int s = 0; s < 2; ++s)
#pragma unroll
      for (int nt = 0; nt < 4; ++nt)
        acc[s][nt] = __builtin_amdgcn_wmma_f32_16x16x32_bf16(
            false, afr[s], false, bfr[nt], (short)0, acc[s][nt], false, false);
  }

  // Epilogue: C layout is VGPR r -> row, lane -> col (lanes 16-31 = rows +8).
#pragma unroll
  for (int s = 0; s < 2; ++s)
#pragma unroll
    for (int nt = 0; nt < 4; ++nt) {
      const int n = n0 + wn * 64 + nt * 16 + l16;
      const float bv = bias[n];
#pragma unroll
      for (int r = 0; r < 8; ++r) {
        const int m = m0 + wm * 32 + s * 16 + half * 8 + r;
        Y[(size_t)m * N + n] = (TOut)(acc[s][nt][r] + bv);
      }
    }
}

// ---------------------------------------------------------------------------
// Flash attention: one workgroup per (b, h, 128-query block); each wave owns
// 16 query rows and streams keys 32 at a time.
// ---------------------------------------------------------------------------
__global__ __launch_bounds__(256)
void attn_kernel(const __bf16* __restrict__ Qb, const __bf16* __restrict__ Kb,
                 const __bf16* __restrict__ Vb, __bf16* __restrict__ Ob) {
  __shared__ alignas(16) __bf16 smK[32 * LDK];       // [key][d]
  __shared__ alignas(16) __bf16 smVt[64 * LDT];      // [d][key] (transposed)
  __shared__ alignas(16) __bf16 smP[8][16 * 32];     // per-wave P re-swizzle

  const int t = threadIdx.x;
  const int lane = t & 31, wave = t >> 5;
  const int half = lane >> 4, l16 = lane & 15;

  int blk = blockIdx.x;
  const int qblk = blk & 7;  blk >>= 3;
  const int h = blk % NHEADS;
  const int b = blk / NHEADS;
  const int q0 = qblk * 128 + wave * 16;
  const size_t baseBH = ((size_t)b * SEQ) * D_MODEL + (size_t)h * DK;
  const __bf16* kbase = Kb + baseBH;

  // Q A-fragments, resident for the whole key loop (d 0..31, 32..63).
  v16bf qa[2];
  {
    const __bf16* qrow = Qb + baseBH + (size_t)(q0 + l16) * D_MODEL;
#pragma unroll
    for (int c = 0; c < 2; ++c)
      qa[c] = join16(*(const v8bf*)(qrow + c * 32 + half * 8),
                     *(const v8bf*)(qrow + c * 32 + 16 + half * 8));
  }
  v16bf ones;
#pragma unroll
  for (int e = 0; e < 16; ++e) ones[e] = (__bf16)1.0f;

  float mrun[8], lrun[8];
  v8f acc[4];
#pragma unroll
  for (int r = 0; r < 8; ++r) { mrun[r] = -1e30f; lrun[r] = 0.0f; }
#pragma unroll
  for (int dt = 0; dt < 4; ++dt)
#pragma unroll
    for (int r = 0; r < 8; ++r) acc[dt][r] = 0.0f;

  for (int kb = 0; kb < SEQ; kb += 32) {
    // --- K tile [32][64] -> LDS, raw bf16 copy: async DMA path. -----------
    {
      const int row = t >> 3, cc = (t & 7) * 8;
#if USE_ASYNC_K
      unsigned ldsoff = (unsigned)(size_t)&smK[row * LDK + cc];
      unsigned voff =
          (unsigned)(((size_t)(kb + row) * D_MODEL + cc) * sizeof(__bf16));
      asm volatile("global_load_async_to_lds_b128 %0, %1, %2 offset:0"
                   :: "v"(ldsoff), "v"(voff), "s"(kbase) : "memory");
#else
      *(v8bf*)&smK[row * LDK + cc] =
          *(const v8bf*)(kbase + (size_t)(kb + row) * D_MODEL + cc);
#endif
    }
    // --- V tile -> LDS transposed [d][key]: 2 keys x 4 d per thread. ------
    {
      const int kp = (t & 15) * 2, d4 = (t >> 4) * 4;
      v4bf va = *(const v4bf*)(Vb + baseBH + (size_t)(kb + kp) * D_MODEL + d4);
      v4bf vb2 = *(const v4bf*)(Vb + baseBH + (size_t)(kb + kp + 1) * D_MODEL + d4);
#pragma unroll
      for (int e = 0; e < 4; ++e) {
        v2bf p; p[0] = va[e]; p[1] = vb2[e];
        *(v2bf*)&smVt[(d4 + e) * LDT + kp] = p;
      }
    }
#if USE_ASYNC_K
    asm volatile("s_wait_asynccnt 0x0" ::: "memory");
#endif
    __syncthreads();

    // --- Scores: two 16-key subtiles, K-dim = d (two chained WMMAs). ------
    v8f sfr[2];
#pragma unroll
    for (int kt = 0; kt < 2; ++kt) {
      v8f s;
#pragma unroll
      for (int r = 0; r < 8; ++r) s[r] = 0.0f;
#pragma unroll
      for (int c = 0; c < 2; ++c) {
        const __bf16* base = &smK[(kt * 16 + l16) * LDK + c * 32];
        v16bf bk = join16(*(const v8bf*)(base + half * 8),
                          *(const v8bf*)(base + 16 + half * 8));
        s = __builtin_amdgcn_wmma_f32_16x16x32_bf16(
            false, qa[c], false, bk, (short)0, s, false, false);
      }
      sfr[kt] = s;
    }

    // --- Online softmax in exp2 domain (scale folded into CSC). -----------
    float corr[8];
#pragma unroll
    for (int r = 0; r < 8; ++r) {
      const float tm = max16(fmaxf(sfr[0][r], sfr[1][r]));  // raw-score max
      const float mn = fmaxf(mrun[r], tm * CSC);
      corr[r] = fast_exp2(mrun[r] - mn);
      mrun[r] = mn;
#pragma unroll
      for (int kt = 0; kt < 2; ++kt)
        sfr[kt][r] = fast_exp2(fmaf(sfr[kt][r], CSC, -mn));
    }
#pragma unroll
    for (int dt = 0; dt < 4; ++dt)
#pragma unroll
      for (int r = 0; r < 8; ++r) acc[dt][r] *= corr[r];

    // --- P: C layout -> A layout via per-wave LDS scratch. ----------------
#pragma unroll
    for (int kt = 0; kt < 2; ++kt)
#pragma unroll
      for (int r = 0; r < 8; ++r)
        smP[wave][(half * 8 + r) * 32 + kt * 16 + l16] = (__bf16)sfr[kt][r];
    v16bf pa;
    {
      const __bf16* base = &smP[wave][l16 * 32];
      pa = join16(*(const v8bf*)(base + half * 8),
                  *(const v8bf*)(base + 16 + half * 8));
    }

    // --- Row sums of P via ones-fragment WMMA (every lane gets its row). --
    v8f ssum;
#pragma unroll
    for (int r = 0; r < 8; ++r) ssum[r] = 0.0f;
    ssum = __builtin_amdgcn_wmma_f32_16x16x32_bf16(
        false, pa, false, ones, (short)0, ssum, false, false);
#pragma unroll
    for (int r = 0; r < 8; ++r) lrun[r] = lrun[r] * corr[r] + ssum[r];

    // --- out += P(16x32) * V(32x64): 4 d-tiles from transposed V. ---------
#pragma unroll
    for (int dt = 0; dt < 4; ++dt) {
      const __bf16* base = &smVt[(dt * 16 + l16) * LDT];
      v16bf bv = join16(*(const v8bf*)(base + half * 8),
                        *(const v8bf*)(base + 16 + half * 8));
      acc[dt] = __builtin_amdgcn_wmma_f32_16x16x32_bf16(
          false, pa, false, bv, (short)0, acc[dt], false, false);
    }
    __syncthreads();
  }

  // Normalize and store concat layout [B,S,H*64] as bf16.
#pragma unroll
  for (int dt = 0; dt < 4; ++dt)
#pragma unroll
    for (int r = 0; r < 8; ++r) {
      const int m = q0 + half * 8 + r;
      const float val = acc[dt][r] / lrun[r];
      Ob[((size_t)b * SEQ + m) * D_MODEL + h * DK + dt * 16 + l16] =
          (__bf16)val;
    }
}

// ---------------------------------------------------------------------------
extern "C" void kernel_launch(void* const* d_in, const int* in_sizes, int n_in,
                              void* d_out, int out_size, void* d_ws,
                              size_t ws_size, hipStream_t stream) {
  (void)in_sizes; (void)n_in; (void)out_size; (void)ws_size;
  const float* key   = (const float*)d_in[0];
  const float* query = (const float*)d_in[1];
  const float* value = (const float*)d_in[2];
  const float* Wk = (const float*)d_in[3];
  const float* bk = (const float*)d_in[4];
  const float* Wq = (const float*)d_in[5];
  const float* bq = (const float*)d_in[6];
  const float* Wv = (const float*)d_in[7];
  const float* bv = (const float*)d_in[8];
  const float* Wo = (const float*)d_in[9];
  const float* bo = (const float*)d_in[10];
  float* out = (float*)d_out;

  const size_t elems = (size_t)MROWS * D_MODEL;
  __bf16* Qb = (__bf16*)d_ws;   // [8192,768] bf16 each
  __bf16* Kb = Qb + elems;
  __bf16* Vb = Kb + elems;
  __bf16* Ab = Vb + elems;      // attention concat output

  dim3 blk(256);
  dim3 gproj(D_MODEL / 128, MROWS / 128);  // (6, 64)
  proj_gemm<float, __bf16><<<gproj, blk, 0, stream>>>(query, Wq, bq, Qb,
                                                      MROWS, D_MODEL, D_MODEL);
  proj_gemm<float, __bf16><<<gproj, blk, 0, stream>>>(key, Wk, bk, Kb,
                                                      MROWS, D_MODEL, D_MODEL);
  proj_gemm<float, __bf16><<<gproj, blk, 0, stream>>>(value, Wv, bv, Vb,
                                                      MROWS, D_MODEL, D_MODEL);
  attn_kernel<<<dim3(NBATCH * NHEADS * (SEQ / 128)), blk, 0, stream>>>(Qb, Kb,
                                                                       Vb, Ab);
  proj_gemm<__bf16, float><<<gproj, blk, 0, stream>>>(Ab, Wo, bo, out,
                                                      MROWS, D_MODEL, D_MODEL);
}
